// mhaModel_6141803233624
// MI455X (gfx1250) — matrix operand order repaired and checked by reading, not compiled, not run on hardware
//
#include <hip/hip_runtime.h>
#include <hip/hip_bf16.h>

// ---------------------------------------------------------------------------
// GPT forward (4 layers, S=2048, D=1024, H=16, HD=64, FF=4096, V=32000)
// All GEMMs + attention matmuls via v_wmma_f32_16x16x32_bf16 (fp32 accum).
// ---------------------------------------------------------------------------

#define S_LEN 2048
#define DMODEL 1024
#define NHEAD 16
#define HDIM 64
#define FFDIM 4096
#define NLAYER 4
#define VOCAB 32000

typedef __bf16 v16bf __attribute__((ext_vector_type(16)));
typedef float  v8f   __attribute__((ext_vector_type(8)));

union FragB16 { v16bf v; uint4 q[2]; };

static __device__ __forceinline__ v8f wmma_bf16(v16bf a, v16bf b, v8f c) {
    return __builtin_amdgcn_wmma_f32_16x16x32_bf16(false, a, false, b,
                                                   (short)0, c, false, false);
}

// ---------------------------------------------------------------------------
// Embedding: x[t, d] = tok_emb[idx[t], d] + pos_emb[t, d]
// ---------------------------------------------------------------------------
__global__ __launch_bounds__(256) void embed_kernel(
    const int* __restrict__ idx, const float* __restrict__ tok,
    const float* __restrict__ pos, float* __restrict__ x)
{
    const int t = blockIdx.x;
    const int token = idx[t];
    const size_t xb = (size_t)t * DMODEL;
    const size_t tb = (size_t)token * DMODEL;
    for (int i = threadIdx.x; i < DMODEL; i += 256)
        x[xb + i] = tok[tb + i] + pos[xb + i];
}

// ---------------------------------------------------------------------------
// LayerNorm (fp32 in, bf16 out): one row of D=1024 per workgroup
// ---------------------------------------------------------------------------
__global__ __launch_bounds__(256) void ln_kernel(
    const float* __restrict__ x, const float* __restrict__ g,
    const float* __restrict__ b, __bf16* __restrict__ out)
{
    const int row = blockIdx.x;
    const float* xr = x + (size_t)row * DMODEL;
    float v[4], s = 0.f, s2 = 0.f;
#pragma unroll
    for (int i = 0; i < 4; ++i) {
        v[i] = xr[threadIdx.x + i * 256];
        s += v[i]; s2 += v[i] * v[i];
    }
#pragma unroll
    for (int o = 16; o; o >>= 1) {
        s  += __shfl_xor(s,  o);
        s2 += __shfl_xor(s2, o);
    }
    __shared__ float sh[16];
    const int wv = threadIdx.x >> 5;
    if ((threadIdx.x & 31) == 0) { sh[wv] = s; sh[wv + 8] = s2; }
    __syncthreads();
    s = 0.f; s2 = 0.f;
#pragma unroll
    for (int w = 0; w < 8; ++w) { s += sh[w]; s2 += sh[w + 8]; }
    const float mean = s * (1.f / DMODEL);
    const float var  = s2 * (1.f / DMODEL) - mean * mean;
    const float rstd = rsqrtf(var + 1e-5f);
#pragma unroll
    for (int i = 0; i < 4; ++i) {
        const int c = threadIdx.x + i * 256;
        out[(size_t)row * DMODEL + c] = (__bf16)((v[i] - mean) * rstd * g[c] + b[c]);
    }
}

// ---------------------------------------------------------------------------
// Split QKV (fp32) -> qb (bf16, pre-scaled by HD^-0.5), kb (bf16), vt (bf16,
// transposed per head: vt[h][hd][s]) so attention V-fragments are contiguous.
// ---------------------------------------------------------------------------
__global__ __launch_bounds__(256) void qkv_split_kernel(
    const float* __restrict__ qkv, __bf16* __restrict__ qb,
    __bf16* __restrict__ kb, __bf16* __restrict__ vt)
{
    const int s = blockIdx.x;
    const size_t base = (size_t)s * (3 * DMODEL);
    for (int d = threadIdx.x; d < DMODEL; d += 256) {
        qb[(size_t)s * DMODEL + d] = (__bf16)(qkv[base + d] * 0.125f); // 1/sqrt(64)
        kb[(size_t)s * DMODEL + d] = (__bf16)(qkv[base + DMODEL + d]);
        const int h = d >> 6, hd = d & 63;
        vt[((size_t)h * HDIM + hd) * S_LEN + s] = (__bf16)(qkv[base + 2 * DMODEL + d]);
    }
}

// ---------------------------------------------------------------------------
// Tiled bf16 GEMM with fp32 accumulate:  C = epi(A[MxK]*B[KxN])
//   A: bf16 row-major (activations), B: fp32 row-major (weights), converted
//   to bf16 while staging to LDS (B staged TRANSPOSED so both A and B
//   fragments are two ds_load_b128 each, both in contiguous k = 16*hlf + s order).
// Epilogue MODE (compile-time, branch-free):
//   0: Cf = acc + bias            1: Cf = acc + bias + res
//   2: Cb = bf16(gelu(acc+bias))  3: Cf = acc           (no bias; head)
// Block tile 128x128, K-step 32, 256 threads = 8 waves (2x4 wave grid),
// each wave computes 64x32 via 8 v_wmma_f32_16x16x32_bf16 accumulators.
// ---------------------------------------------------------------------------
#define BM 128
#define BN 128
#define BK 32

template<int MODE>
__global__ __launch_bounds__(256) void gemm_bf16_kernel(
    const __bf16* __restrict__ A, const float* __restrict__ B,
    const float* __restrict__ bias, const float* __restrict__ res,
    float* __restrict__ Cf, __bf16* __restrict__ Cb,
    int M, int N, int K)
{
    __shared__ __align__(32) __bf16 As [BM][BK + 8];  // 40 halfs/row = 80B (16B-mult)
    __shared__ __align__(32) __bf16 Bst[BN][BK + 8];  // transposed: [n][k]

    const int tid  = threadIdx.x;
    const int lane = tid & 31;
    const int wv   = tid >> 5;     // 0..7
    const int wm   = wv >> 2;      // 0..1  (M half of block)
    const int wn   = wv & 3;       // 0..3  (N quarter of block)
    const int hlf  = lane >> 4;    // 0/1
    const int l16  = lane & 15;

    const int bm0 = blockIdx.y * BM;
    const int bn0 = blockIdx.x * BN;

    v8f acc[4][2];
#pragma unroll
    for (int mt = 0; mt < 4; ++mt)
#pragma unroll
        for (int nt = 0; nt < 2; ++nt)
#pragma unroll
            for (int i = 0; i < 8; ++i) acc[mt][nt][i] = 0.f;

    for (int k0 = 0; k0 < K; k0 += BK) {
        __syncthreads();   // previous-iteration LDS reads complete
        // ---- stage A: 128x32 bf16, 16B chunks (8 halfs) ----
#pragma unroll
        for (int r = 0; r < 2; ++r) {
            const int c   = tid + r * 256;       // 512 chunks total
            const int row = c >> 2;              // 4 chunks per row
            const int off = (c & 3) * 8;
            *(uint4*)(&As[row][off]) =
                *(const uint4*)(A + (size_t)(bm0 + row) * K + k0 + off);
        }
        // ---- stage B transposed: fp32 -> bf16, (k,k+1) pair-packed u32 ----
        //      consecutive tid -> consecutive n  => coalesced global b32 loads
#pragma unroll
        for (int r = 0; r < 8; ++r) {
            const int p  = tid + r * 256;        // 2048 (n, k-pair) items
            const int n  = p & 127;
            const int k2 = (p >> 7) * 2;         // 0,2,...,30
            const float f0 = B[(size_t)(k0 + k2)     * N + bn0 + n];
            const float f1 = B[(size_t)(k0 + k2 + 1) * N + bn0 + n];
            union { __bf16 h[2]; unsigned u; } pk;
            pk.h[0] = (__bf16)f0; pk.h[1] = (__bf16)f1;
            *(unsigned*)(&Bst[n][k2]) = pk.u;
        }
        __syncthreads();

        // ---- B fragments: K contiguous per lane -> 2x ds_load_b128 each ----
        FragB16 b0f, b1f;
        const int n0 = wn * 32 + l16;
        b0f.q[0] = *(const uint4*)(&Bst[n0][hlf * 16]);
        b0f.q[1] = *(const uint4*)(&Bst[n0][hlf * 16 + 8]);
        b1f.q[0] = *(const uint4*)(&Bst[n0 + 16][hlf * 16]);
        b1f.q[1] = *(const uint4*)(&Bst[n0 + 16][hlf * 16 + 8]);
        // ---- A fragments (k = 16*hlf + s, same order as B) + 8 WMMAs ----
#pragma unroll
        for (int mt = 0; mt < 4; ++mt) {
            const int row = wm * 64 + mt * 16 + l16;
            FragB16 af;
            af.q[0] = *(const uint4*)(&As[row][hlf * 16]);       // K = hlf*16 .. +7
            af.q[1] = *(const uint4*)(&As[row][hlf * 16 + 8]);   // K = hlf*16+8 .. +15
            acc[mt][0] = wmma_bf16(af.v, b0f.v, acc[mt][0]);
            acc[mt][1] = wmma_bf16(af.v, b1f.v, acc[mt][1]);
        }
    }

    // ---- epilogue: C/D layout is c[r] -> (m = r + 8*hlf, n = l16) ----
#pragma unroll
    for (int mt = 0; mt < 4; ++mt)
#pragma unroll
        for (int nt = 0; nt < 2; ++nt) {
            const int col = bn0 + wn * 32 + nt * 16 + l16;
            const float bv = (MODE == 3) ? 0.f : bias[col];
#pragma unroll
            for (int r = 0; r < 8; ++r) {
                const int rowg = bm0 + wm * 64 + mt * 16 + 8 * hlf + r;
                const size_t ci = (size_t)rowg * N + col;
                float v = acc[mt][nt][r] + bv;
                if (MODE == 2) {
                    v = 0.5f * v * (1.f + erff(v * 0.70710678118654752f));
                    Cb[ci] = (__bf16)v;
                } else if (MODE == 1) {
                    Cf[ci] = v + res[ci];
                } else {
                    Cf[ci] = v;
                }
            }
        }
}

// ---------------------------------------------------------------------------
// Flash attention, causal. Grid: (S/64, H). 128 threads = 4 waves; each wave
// owns 16 query rows. Per 32-key block: Q*K^T (4 WMMA), fp32 online softmax
// (16-lane shfl_xor row reductions), P through LDS (layout D->A), P*V (4 WMMA).
// Uniform jb loop across waves (masking handles causality) -> barriers legal.
// ---------------------------------------------------------------------------
__global__ __launch_bounds__(128) void attn_kernel(
    const __bf16* __restrict__ qb, const __bf16* __restrict__ kb,
    const __bf16* __restrict__ vt, __bf16* __restrict__ ao)
{
    __shared__ __align__(32) __bf16 Ks[32][72];       // [key][hd], padded
    __shared__ __align__(32) __bf16 Vs[64][40];       // [hd][key], padded
    __shared__ __align__(32) __bf16 Ps[4][16][40];    // per-wave P tile [m][key]

    const int tid  = threadIdx.x;
    const int lane = tid & 31;
    const int wv   = tid >> 5;       // 0..3
    const int hlf  = lane >> 4;
    const int l16  = lane & 15;
    const int h    = blockIdx.y;
    const int qb0  = blockIdx.x * 64;
    const int qrow0 = qb0 + wv * 16;

    // Q fragments (2 K-steps over HD=64), loaded straight from global
    v16bf qf[2];
    {
        const __bf16* qp = qb + (size_t)(qrow0 + l16) * DMODEL + h * HDIM;
#pragma unroll
        for (int ks = 0; ks < 2; ++ks) {
            FragB16 f;
            f.q[0] = *(const uint4*)(qp + ks * 32 + hlf * 16);
            f.q[1] = *(const uint4*)(qp + ks * 32 + hlf * 16 + 8);
            qf[ks] = f.v;
        }
    }

    float mrow[8], lrow[8];
    v8f o[4];
#pragma unroll
    for (int r = 0; r < 8; ++r) { mrow[r] = -1e30f; lrow[r] = 0.f; }
#pragma unroll
    for (int t = 0; t < 4; ++t)
#pragma unroll
        for (int i = 0; i < 8; ++i) o[t][i] = 0.f;

    const int kend = qb0 + 64;   // workgroup causal bound (uniform)
    for (int jb = 0; jb < kend; jb += 32) {
        __syncthreads();   // previous-iteration LDS reads complete
        // ---- stage K tile [32 keys][64 hd] and V^T tile [64 hd][32 keys] ----
#pragma unroll
        for (int r = 0; r < 2; ++r) {
            const int c   = tid + r * 128;       // 256 chunks of 8 halfs
            const int key = c >> 3;
            const int off = (c & 7) * 8;
            *(uint4*)(&Ks[key][off]) =
                *(const uint4*)(kb + (size_t)(jb + key) * DMODEL + h * HDIM + off);
        }
#pragma unroll
        for (int r = 0; r < 2; ++r) {
            const int c   = tid + r * 128;
            const int hd  = c >> 2;              // 4 chunks per hd row
            const int off = (c & 3) * 8;
            *(uint4*)(&Vs[hd][off]) =
                *(const uint4*)(vt + ((size_t)h * HDIM + hd) * S_LEN + jb + off);
        }
        __syncthreads();

        // ---- scores for two 16-key subtiles ----
        v8f sc0, sc1;
#pragma unroll
        for (int i = 0; i < 8; ++i) { sc0[i] = 0.f; sc1[i] = 0.f; }
#pragma unroll
        for (int ks = 0; ks < 2; ++ks) {
            FragB16 k0f, k1f;
            k0f.q[0] = *(const uint4*)(&Ks[l16][ks * 32 + hlf * 16]);
            k0f.q[1] = *(const uint4*)(&Ks[l16][ks * 32 + hlf * 16 + 8]);
            k1f.q[0] = *(const uint4*)(&Ks[16 + l16][ks * 32 + hlf * 16]);
            k1f.q[1] = *(const uint4*)(&Ks[16 + l16][ks * 32 + hlf * 16 + 8]);
            sc0 = wmma_bf16(qf[ks], k0f.v, sc0);
            sc1 = wmma_bf16(qf[ks], k1f.v, sc1);
        }

        // ---- causal mask + online softmax (rows live in 16-lane halves) ----
        float mnew[8];
#pragma unroll
        for (int r = 0; r < 8; ++r) {
            const int qrow = qrow0 + 8 * hlf + r;
            const float a0 = (jb + l16      <= qrow) ? sc0[r] : -1e30f;
            const float a1 = (jb + 16 + l16 <= qrow) ? sc1[r] : -1e30f;
            sc0[r] = a0; sc1[r] = a1;
            float v = fmaxf(a0, a1);
#pragma unroll
            for (int off = 8; off; off >>= 1) v = fmaxf(v, __shfl_xor(v, off));
            mnew[r] = fmaxf(mrow[r], v);
        }
#pragma unroll
        for (int r = 0; r < 8; ++r) {
            float p0, p1, alpha;
            if (mnew[r] <= -1e29f) { p0 = 0.f; p1 = 0.f; alpha = 1.f; }
            else {
                alpha = __expf(mrow[r] - mnew[r]);
                p0 = (sc0[r] <= -1e29f) ? 0.f : __expf(sc0[r] - mnew[r]);
                p1 = (sc1[r] <= -1e29f) ? 0.f : __expf(sc1[r] - mnew[r]);
            }
            float ps = p0 + p1;
#pragma unroll
            for (int off = 8; off; off >>= 1) ps += __shfl_xor(ps, off);
            lrow[r] = lrow[r] * alpha + ps;
            mrow[r] = mnew[r];
#pragma unroll
            for (int t = 0; t < 4; ++t) o[t][r] *= alpha;
            const int m = 8 * hlf + r;
            Ps[wv][m][l16]      = (__bf16)p0;
            Ps[wv][m][16 + l16] = (__bf16)p1;
        }
        __syncthreads();   // P store -> P reload ordering (uniform)

        // ---- P (A-fragment, 16x32) x V (B-fragment from Vs) ----
        FragB16 pf;
        pf.q[0] = *(const uint4*)(&Ps[wv][l16][hlf * 16]);
        pf.q[1] = *(const uint4*)(&Ps[wv][l16][hlf * 16 + 8]);
#pragma unroll
        for (int t = 0; t < 4; ++t) {
            FragB16 vf;
            vf.q[0] = *(const uint4*)(&Vs[t * 16 + l16][hlf * 16]);
            vf.q[1] = *(const uint4*)(&Vs[t * 16 + l16][hlf * 16 + 8]);
            o[t] = wmma_bf16(pf.v, vf.v, o[t]);
        }
    }

    // ---- normalize and emit bf16 (input to out-proj GEMM) ----
#pragma unroll
    for (int r = 0; r < 8; ++r) lrow[r] = 1.f / lrow[r];
#pragma unroll
    for (int t = 0; t < 4; ++t)
#pragma unroll
        for (int r = 0; r < 8; ++r) {
            const int row = qrow0 + 8 * hlf + r;
            const int col = h * HDIM + t * 16 + l16;
            ao[(size_t)row * DMODEL + col] = (__bf16)(o[t][r] * lrow[r]);
        }
}

// ---------------------------------------------------------------------------
extern "C" void kernel_launch(void* const* d_in, const int* in_sizes, int n_in,
                              void* d_out, int out_size, void* d_ws, size_t ws_size,
                              hipStream_t stream)
{
    (void)in_sizes; (void)n_in; (void)out_size; (void)ws_size;
    const int*   idx     = (const int*)  d_in[0];
    const float* tok_emb = (const float*)d_in[1];
    const float* pos_emb = (const float*)d_in[2];
    const float* qkv_w   = (const float*)d_in[3];
    const float* qkv_b   = (const float*)d_in[4];
    const float* out_w   = (const float*)d_in[5];
    const float* out_b   = (const float*)d_in[6];
    const float* ln1_g   = (const float*)d_in[7];
    const float* ln1_b   = (const float*)d_in[8];
    const float* ln2_g   = (const float*)d_in[9];
    const float* ln2_b   = (const float*)d_in[10];
    const float* ff1_w   = (const float*)d_in[11];
    const float* ff1_b   = (const float*)d_in[12];
    const float* ff2_w   = (const float*)d_in[13];
    const float* ff2_b   = (const float*)d_in[14];
    const float* lnf_g   = (const float*)d_in[15];
    const float* lnf_b   = (const float*)d_in[16];
    const float* head_w  = (const float*)d_in[17];
    float* logits = (float*)d_out;

    // workspace carve (~68 MiB, all 4KB-aligned sizes)
    char* ws = (char*)d_ws;
    float*  x    = (float*)ws;  ws += (size_t)S_LEN * DMODEL * 4;       // 8 MiB
    float*  qkv  = (float*)ws;  ws += (size_t)S_LEN * 3 * DMODEL * 4;   // 24 MiB
    __bf16* xn   = (__bf16*)ws; ws += (size_t)S_LEN * DMODEL * 2;       // 4 MiB
    __bf16* qb   = (__bf16*)ws; ws += (size_t)S_LEN * DMODEL * 2;
    __bf16* kbuf = (__bf16*)ws; ws += (size_t)S_LEN * DMODEL * 2;
    __bf16* vt   = (__bf16*)ws; ws += (size_t)S_LEN * DMODEL * 2;
    __bf16* ao   = (__bf16*)ws; ws += (size_t)S_LEN * DMODEL * 2;
    __bf16* hbuf = (__bf16*)ws; ws += (size_t)S_LEN * FFDIM * 2;        // 16 MiB

    const dim3 gq (3 * DMODEL / BN, S_LEN / BM);   // 24 x 16
    const dim3 go (DMODEL / BN,     S_LEN / BM);   //  8 x 16
    const dim3 gf1(FFDIM / BN,      S_LEN / BM);   // 32 x 16
    const dim3 gf2(DMODEL / BN,     S_LEN / BM);
    const dim3 gh (VOCAB / BN,      S_LEN / BM);   // 250 x 16
    const dim3 ga (S_LEN / 64, NHEAD);             // 32 x 16

    embed_kernel<<<S_LEN, 256, 0, stream>>>(idx, tok_emb, pos_emb, x);

    for (int l = 0; l < NLAYER; ++l) {
        ln_kernel<<<S_LEN, 256, 0, stream>>>(x, ln1_g + (size_t)l * DMODEL,
                                             ln1_b + (size_t)l * DMODEL, xn);
        gemm_bf16_kernel<0><<<gq, 256, 0, stream>>>(
            xn, qkv_w + (size_t)l * DMODEL * 3 * DMODEL, qkv_b + (size_t)l * 3 * DMODEL,
            nullptr, qkv, nullptr, S_LEN, 3 * DMODEL, DMODEL);
        qkv_split_kernel<<<S_LEN, 256, 0, stream>>>(qkv, qb, kbuf, vt);
        attn_kernel<<<ga, 128, 0, stream>>>(qb, kbuf, vt, ao);
        gemm_bf16_kernel<1><<<go, 256, 0, stream>>>(
            ao, out_w + (size_t)l * DMODEL * DMODEL, out_b + (size_t)l * DMODEL,
            x, x, nullptr, S_LEN, DMODEL, DMODEL);             // += residual, in-place
        ln_kernel<<<S_LEN, 256, 0, stream>>>(x, ln2_g + (size_t)l * DMODEL,
                                             ln2_b + (size_t)l * DMODEL, xn);
        gemm_bf16_kernel<2><<<gf1, 256, 0, stream>>>(
            xn, ff1_w + (size_t)l * DMODEL * FFDIM, ff1_b + (size_t)l * FFDIM,
            nullptr, nullptr, hbuf, S_LEN, FFDIM, DMODEL);     // exact GELU, bf16 out
        gemm_bf16_kernel<1><<<gf2, 256, 0, stream>>>(
            hbuf, ff2_w + (size_t)l * FFDIM * DMODEL, ff2_b + (size_t)l * DMODEL,
            x, x, nullptr, S_LEN, DMODEL, FFDIM);              // += residual
    }

    ln_kernel<<<S_LEN, 256, 0, stream>>>(x, lnf_g, lnf_b, xn);
    gemm_bf16_kernel<3><<<gh, 256, 0, stream>>>(
        xn, head_w, nullptr, nullptr, logits, nullptr, S_LEN, VOCAB, DMODEL);
}
